// NodeAttentionLayer_6133213299381
// MI455X (gfx1250) — compile-verified, matrix operand order checked by reading, not run on hardware
//
#include <hip/hip_runtime.h>
#include <hip/hip_fp16.h>

#define NN   100000
#define NE   800000
#define INND 128
#define OUTD 128
#define NH   8
#define HD   16
#define QCOLS (3*OUTD)   // 384

typedef __attribute__((ext_vector_type(16))) _Float16 v16h;
typedef __attribute__((ext_vector_type(8)))  float    v8f;
typedef __attribute__((ext_vector_type(4)))  unsigned int v4u;
typedef __attribute__((ext_vector_type(8)))  int      v8i;
typedef __attribute__((ext_vector_type(4)))  int      v4i;

// ---------------- WMMA fragment helpers (16-bit A/B layout, wave32) ----------
// lane l: index (l&15), K pattern: lanes 0-15 -> K = {0..7, 16..23},
//                                  lanes16-31 -> K = {8..15, 24..31}
__device__ __forceinline__ v16h frag_from_f32(const float* __restrict__ row, int k0, int kbase) {
  v16h a;
#pragma unroll
  for (int i = 0; i < 8; ++i) a[i]     = (_Float16)row[k0 + kbase + i];
#pragma unroll
  for (int i = 0; i < 8; ++i) a[8 + i] = (_Float16)row[k0 + 16 + kbase + i];
  return a;
}
__device__ __forceinline__ v16h frag_from_f16(const _Float16* __restrict__ row, int k0, int kbase) {
  v16h a;
#pragma unroll
  for (int i = 0; i < 8; ++i) a[i]     = row[k0 + kbase + i];
#pragma unroll
  for (int i = 0; i < 8; ++i) a[8 + i] = row[k0 + 16 + kbase + i];
  return a;
}

// ---------------- TDM: DMA a contiguous run of f16 into LDS ------------------
// Builds a D# (group0/group1 per CDNA5 ISA 8.3/8.4) describing a 1-row tile of
// n_elems 2-byte elements and issues tensor_load_to_lds (TENSORcnt-tracked).
// amdgpu-toolchain builtin arity: (v4u, v8i, v4i, v4i, v8i, i32 cpol).
__device__ __forceinline__ void tdm_load_f16(const _Float16* gsrc, _Float16* lds_dst, int n_elems) {
  unsigned long long ga = (unsigned long long)(uintptr_t)gsrc;
  unsigned int       la = (unsigned int)(uintptr_t)lds_dst;   // LDS aperture: addr[31:0] = LDS offset
  v4u g0;
  g0[0] = 1u;                                        // count=1, user descriptor, no gather
  g0[1] = la;                                        // lds_addr (bytes)
  g0[2] = (unsigned int)(ga & 0xFFFFFFFFu);          // global_addr[31:0]
  g0[3] = (unsigned int)((ga >> 32) & 0x01FFFFFFu)   // global_addr[56:32]
        | (2u << 30);                                // type = 2 ("image")
  v8i g1;
  unsigned int ne = (unsigned int)n_elems;
  g1[0] = (int)(1u << 16);                           // data_size = 1 (2 bytes); no multicast/pad
  g1[1] = (int)((ne & 0xFFFFu) << 16);               // tensor_dim0[15:0]
  g1[2] = (int)((ne >> 16) | (1u << 16));            // tensor_dim0[31:16] | tensor_dim1 = 1
  g1[3] = (int)((ne & 0xFFFFu) << 16);               // tile_dim0 = n_elems
  g1[4] = 1;                                         // tile_dim1 = 1, tile_dim2 = 0
  g1[5] = (int)ne;                                   // tensor_dim0_stride[31:0]
  g1[6] = 0;                                         // stride0[47:32] | stride1[15:0]
  g1[7] = 0;                                         // stride1[47:16]
  v4i g2 = {0, 0, 0, 0};
  v4i g3 = {0, 0, 0, 0};
  v8i g4 = {0, 0, 0, 0, 0, 0, 0, 0};
  __builtin_amdgcn_tensor_load_to_lds(g0, g1, g2, g3, g4, 0);
}

__device__ __forceinline__ void atomicMaxF(float* addr, float val) {
  if (val >= 0.0f) atomicMax((int*)addr, __float_as_int(val));
  else             atomicMin((unsigned int*)addr, __float_as_uint(val));
}

// ---------------- Kernel 0: convert weights to f16 ---------------------------
__global__ __launch_bounds__(256) void cvt_weights(const float* __restrict__ Wqkv,
                                                   const float* __restrict__ Wprj,
                                                   _Float16* __restrict__ Wqkvh,
                                                   _Float16* __restrict__ Wprjh) {
  int idx = blockIdx.x * blockDim.x + threadIdx.x;
  if (idx < QCOLS * INND) Wqkvh[idx] = (_Float16)Wqkv[idx];
  int p = idx - QCOLS * INND;
  if (p >= 0 && p < OUTD * OUTD) Wprjh[p] = (_Float16)Wprj[p];
}

// ---------------- Kernel 1: init softmax stats + f16 accumulator -------------
__global__ __launch_bounds__(256) void init_state(float* __restrict__ m,
                                                  float* __restrict__ denom,
                                                  _Float16* __restrict__ acc) {
  int idx = blockIdx.x * blockDim.x + threadIdx.x;
  if (idx < NN * NH) { m[idx] = -__builtin_inff(); denom[idx] = 0.0f; }
  if (idx < NN * OUTD) acc[idx] = (_Float16)0.0f;
}

// ---------------- Kernel 2: QKV GEMM via v_wmma_f32_16x16x32_f16 -------------
// 8 waves/block share one 16-col weight tile: TDM it into LDS once, then each
// wave computes a 16x16 output tile. Cols 0..383 scatter into q/k/v (f16).
__global__ __launch_bounds__(256) void qkv_gemm_wmma(const float* __restrict__ x,
                                                     const _Float16* __restrict__ Wh,
                                                     _Float16* __restrict__ qh,
                                                     _Float16* __restrict__ kh,
                                                     _Float16* __restrict__ vh) {
  __shared__ __align__(16) _Float16 bsm[16 * INND];     // 4 KB weight tile
  const int colTile = blockIdx.y;                       // 0..23
  if (threadIdx.x < 32) {                               // wave 0 issues the DMA
    tdm_load_f16(Wh + (size_t)colTile * 16 * INND, bsm, 16 * INND);
    __builtin_amdgcn_s_wait_tensorcnt(0);
  }
  __syncthreads();

  const int lane    = threadIdx.x & 31;
  const int rowTile = blockIdx.x * 8 + (threadIdx.x >> 5);
  if (rowTile >= NN / 16) return;                       // whole-wave uniform exit
  const int l16   = lane & 15;
  const int kbase = (lane >> 4) << 3;                   // 0 or 8

  const float*    xrow = x + (size_t)(rowTile * 16 + l16) * INND;
  const _Float16* wrow = bsm + (size_t)l16 * INND;      // B fragment from LDS

  v8f c = {};
#pragma unroll
  for (int k0 = 0; k0 < INND; k0 += 32) {
    v16h a = frag_from_f32(xrow, k0, kbase);
    v16h b = frag_from_f16(wrow, k0, kbase);
    c = __builtin_amdgcn_wmma_f32_16x16x32_f16(false, a, false, b, (short)0, c, false, false);
  }

  // D layout: VGPR i -> M = i (lanes 0-15) or 8+i (lanes 16-31), N = lane&15
  const int col  = colTile * 16 + l16;
  const int head = col / (3 * HD);
  const int rem  = col % (3 * HD);
  _Float16* dst; int dc;
  if (rem < HD)            { dst = qh; dc = rem; }
  else if (rem < 2 * HD)   { dst = kh; dc = rem - HD; }
  else                     { dst = vh; dc = rem - 2 * HD; }
  const int mOff = (lane >> 4) << 3;
#pragma unroll
  for (int i = 0; i < 8; ++i) {
    int row = rowTile * 16 + mOff + i;
    dst[(size_t)row * (NH * HD) + head * HD + dc] = (_Float16)c[i];
  }
}

// ---------------- Kernel 3: alpha = q.k/4 + segment max ----------------------
__global__ __launch_bounds__(256) void edge_alpha_max(const _Float16* __restrict__ qh,
                                                      const _Float16* __restrict__ kh,
                                                      const int* __restrict__ inode,
                                                      const int* __restrict__ jnode,
                                                      float* __restrict__ alpha,
                                                      float* __restrict__ mbuf) {
  size_t t = (size_t)blockIdx.x * blockDim.x + threadIdx.x;
  if (t >= (size_t)NE * NH) return;
  int e = (int)(t >> 3), h = (int)(t & 7);
  int i = inode[e], j = jnode[e];
  const _Float16* qp = qh + (size_t)i * (NH * HD) + h * HD;
  const _Float16* kp = kh + (size_t)j * (NH * HD) + h * HD;
  float s = 0.0f;
#pragma unroll
  for (int d = 0; d < HD; ++d) s += (float)qp[d] * (float)kp[d];
  float a = s * 0.25f;                              // / SCALE, SCALE = 4
  alpha[t] = a;
  atomicMaxF(mbuf + (size_t)i * NH + h, a);
}

// ---------------- Kernel 4: m = finite ? m : 0 -------------------------------
__global__ __launch_bounds__(256) void m_fixup(float* __restrict__ m) {
  int idx = blockIdx.x * blockDim.x + threadIdx.x;
  if (idx < NN * NH) {
    float v = m[idx];
    if (!__builtin_isfinite(v)) m[idx] = 0.0f;
  }
}

// ---------------- Kernel 5: ex = exp(alpha - m[i]); denom += ex --------------
__global__ __launch_bounds__(256) void edge_exp_sum(float* __restrict__ alpha,
                                                    const float* __restrict__ mbuf,
                                                    float* __restrict__ denom,
                                                    const int* __restrict__ inode) {
  size_t t = (size_t)blockIdx.x * blockDim.x + threadIdx.x;
  if (t >= (size_t)NE * NH) return;
  int e = (int)(t >> 3), h = (int)(t & 7);
  int i = inode[e];
  float ex = __expf(alpha[t] - mbuf[(size_t)i * NH + h]);
  alpha[t] = ex;
  atomicAdd(denom + (size_t)i * NH + h, ex);
}

// ---------------- Kernel 6: acc[i] += w * v[j]  (packed f16 atomics) ---------
__global__ __launch_bounds__(256) void edge_scatter_v(const float* __restrict__ alpha,
                                                      const float* __restrict__ denom,
                                                      const _Float16* __restrict__ vh,
                                                      const int* __restrict__ inode,
                                                      const int* __restrict__ jnode,
                                                      _Float16* __restrict__ acc) {
  size_t t = (size_t)blockIdx.x * blockDim.x + threadIdx.x;
  if (t >= (size_t)NE * NH) return;
  int e = (int)(t >> 3), h = (int)(t & 7);
  int i = inode[e], j = jnode[e];
  float w = alpha[t] / (denom[(size_t)i * NH + h] + 1e-16f);
  const _Float16* vp = vh + (size_t)j * (NH * HD) + h * HD;
  __half2* ap = (__half2*)(acc + (size_t)i * (NH * HD) + h * HD);
#pragma unroll
  for (int d = 0; d < HD / 2; ++d) {
    atomicAdd(ap + d, __floats2half2_rn((float)vp[2 * d] * w, (float)vp[2 * d + 1] * w));
  }
}

// ---------------- Kernel 7: projection GEMM via WMMA (TDM-staged B) ----------
__global__ __launch_bounds__(256) void proj_gemm_wmma(const _Float16* __restrict__ acc,
                                                      const _Float16* __restrict__ Wh,
                                                      float* __restrict__ out) {
  __shared__ __align__(16) _Float16 bsm[16 * OUTD];     // 4 KB weight tile
  const int colTile = blockIdx.y;                       // 0..7
  if (threadIdx.x < 32) {
    tdm_load_f16(Wh + (size_t)colTile * 16 * OUTD, bsm, 16 * OUTD);
    __builtin_amdgcn_s_wait_tensorcnt(0);
  }
  __syncthreads();

  const int lane    = threadIdx.x & 31;
  const int rowTile = blockIdx.x * 8 + (threadIdx.x >> 5);
  if (rowTile >= NN / 16) return;
  const int l16   = lane & 15;
  const int kbase = (lane >> 4) << 3;

  const _Float16* arow = acc + (size_t)(rowTile * 16 + l16) * OUTD;
  const _Float16* wrow = bsm + (size_t)l16 * OUTD;

  v8f c = {};
#pragma unroll
  for (int k0 = 0; k0 < OUTD; k0 += 32) {
    v16h a = frag_from_f16(arow, k0, kbase);
    v16h b = frag_from_f16(wrow, k0, kbase);
    c = __builtin_amdgcn_wmma_f32_16x16x32_f16(false, a, false, b, (short)0, c, false, false);
  }
  const int mOff = (lane >> 4) << 3;
#pragma unroll
  for (int i = 0; i < 8; ++i) {
    int row = rowTile * 16 + mOff + i;
    out[(size_t)row * OUTD + colTile * 16 + l16] = c[i];
  }
}

// ---------------- launcher ---------------------------------------------------
extern "C" void kernel_launch(void* const* d_in, const int* in_sizes, int n_in,
                              void* d_out, int out_size, void* d_ws, size_t ws_size,
                              hipStream_t stream) {
  const float* x     = (const float*)d_in[0];
  const float* Wqkv  = (const float*)d_in[1];
  const float* Wprj  = (const float*)d_in[2];
  const int*   inode = (const int*)d_in[3];
  const int*   jnode = (const int*)d_in[4];
  float*       out   = (float*)d_out;

  // carve workspace (all offsets 256B aligned)
  char* p = (char*)d_ws;
  auto carve = [&](size_t bytes) -> char* {
    char* r = p;
    p += (bytes + 255) & ~(size_t)255;
    return r;
  };
  _Float16* qh    = (_Float16*)carve((size_t)NN * NH * HD * 2);   // 25.6 MB
  _Float16* kh    = (_Float16*)carve((size_t)NN * NH * HD * 2);   // 25.6 MB
  _Float16* vh    = (_Float16*)carve((size_t)NN * NH * HD * 2);   // 25.6 MB
  _Float16* acc   = (_Float16*)carve((size_t)NN * OUTD * 2);      // 25.6 MB
  float*    alpha = (float*)   carve((size_t)NE * NH * 4);        // 25.6 MB
  float*    mbuf  = (float*)   carve((size_t)NN * NH * 4);        //  3.2 MB
  float*    denom = (float*)   carve((size_t)NN * NH * 4);        //  3.2 MB
  _Float16* Wqkvh = (_Float16*)carve((size_t)QCOLS * INND * 2);
  _Float16* Wprjh = (_Float16*)carve((size_t)OUTD * OUTD * 2);
  if ((size_t)(p - (char*)d_ws) > ws_size) return;  // workspace too small

  const int edgeThreads = NE * NH;                  // 6.4M
  const int edgeBlocks  = (edgeThreads + 255) / 256;
  const int rowTiles    = NN / 16;                  // 6250
  const int gemmBlocksX = (rowTiles + 7) / 8;       // 8 waves (tiles) per block

  cvt_weights<<<((QCOLS + OUTD) * INND + 255) / 256 + 64, 256, 0, stream>>>(Wqkv, Wprj, Wqkvh, Wprjh);
  init_state<<<(NN * OUTD + 255) / 256, 256, 0, stream>>>(mbuf, denom, acc);
  qkv_gemm_wmma<<<dim3(gemmBlocksX, QCOLS / 16), 256, 0, stream>>>(x, Wqkvh, qh, kh, vh);
  edge_alpha_max<<<edgeBlocks, 256, 0, stream>>>(qh, kh, inode, jnode, alpha, mbuf);
  m_fixup<<<(NN * NH + 255) / 256, 256, 0, stream>>>(mbuf);
  edge_exp_sum<<<edgeBlocks, 256, 0, stream>>>(alpha, mbuf, denom, inode);
  edge_scatter_v<<<edgeBlocks, 256, 0, stream>>>(alpha, denom, vh, inode, jnode, acc);
  proj_gemm_wmma<<<dim3(gemmBlocksX, OUTD / 16), 256, 0, stream>>>(acc, Wprjh, out);
}
